// Model_41360535060549
// MI455X (gfx1250) — matrix operand order repaired and checked by reading, not compiled, not run on hardware
//
#include <hip/hip_runtime.h>
#include <math.h>

// ---------------------------------------------------------------------------
// Types for CDNA5 WMMA (gfx1250, wave32)
// ---------------------------------------------------------------------------
typedef __attribute__((ext_vector_type(16))) __bf16 v16bf;
typedef __attribute__((ext_vector_type(8)))  __bf16 v8bf;
typedef __attribute__((ext_vector_type(4)))  __bf16 v4bf;
typedef __attribute__((ext_vector_type(8)))  float  v8f;

#define BATCH   32
#define T_IN    336
#define DM      512
#define DFF     2048
#define DB      128
#define NH      8
#define S_VALID 446
#define SPAD    448
#define PREDOUT 672   // 96 * 7

__device__ __forceinline__ float wave_sum(float x) {
#pragma unroll
  for (int o = 16; o > 0; o >>= 1) x += __shfl_xor(x, o, 32);
  return x;
}
__device__ __forceinline__ float wave_max(float x) {
#pragma unroll
  for (int o = 16; o > 0; o >>= 1) x = fmaxf(x, __shfl_xor(x, o, 32));
  return x;
}
__device__ __forceinline__ v16bf cat8(v8bf a, v8bf b) {
  return __builtin_shufflevector(a, b, 0, 1, 2, 3, 4, 5, 6, 7,
                                 8, 9, 10, 11, 12, 13, 14, 15);
}

// ---------------------------------------------------------------------------
// Elementwise f32 -> bf16 (4 elements / thread, vectorized)
// ---------------------------------------------------------------------------
__global__ __launch_bounds__(256) void cvt_bf16_kernel(
    const float* __restrict__ src, __bf16* __restrict__ dst, int n4)
{
  const int i = blockIdx.x * 256 + threadIdx.x;
  if (i >= n4) return;
  const float4 v = ((const float4*)src)[i];
  v4bf o;
  o[0] = (__bf16)v.x; o[1] = (__bf16)v.y; o[2] = (__bf16)v.z; o[3] = (__bf16)v.w;
  ((v4bf*)dst)[i] = o;
}

// ---------------------------------------------------------------------------
// bf16 WMMA GEMM:  C = act(A[MxK]_bf16 * B[KxN]_bf16 + bias + resid)
// block = (32,4): 4 waves, each wave computes a 64x32 strip (4 M-tiles x 2 N-tiles)
// grid  = (N/128, M/64).  M%64==0, N%128==0, K%32==0 (guaranteed by caller).
// act : 0 = none, 1 = exact GELU
// ofmt: 0 = f32, 1 = bf16, 2 = bf16 transposed per batch (K^T layout, N==512)
// ---------------------------------------------------------------------------
__global__ __launch_bounds__(128) void gemm_bf16_kernel(
    const __bf16* __restrict__ A, const __bf16* __restrict__ Bw,
    const float* __restrict__ bias, const float* __restrict__ resid,
    void* __restrict__ Cout, int M, int N, int K, int act, int ofmt)
{
  const int lane = threadIdx.x;
  const int n0   = (blockIdx.x * 4 + threadIdx.y) * 32;
  const int m0   = blockIdx.y * 64;
  const int hl   = lane >> 4;     // lane half (0/1)
  const int l15  = lane & 15;

  v8f acc[4][2] = {};

  for (int kk = 0; kk < K; kk += 32) {
    // B fragments: lane holds column l15 of B; k in table order.
    const __bf16* bp = Bw + (size_t)kk * N + n0 + l15;
    v16bf bf0, bf1;
#pragma unroll
    for (int e = 0; e < 16; ++e) {
      const int kr = (e < 8) ? (hl * 8 + e) : (16 + hl * 8 + e - 8);
      bf0[e] = bp[(size_t)kr * N];
      bf1[e] = bp[(size_t)kr * N + 16];
    }
    if (kk + 32 < K) __builtin_prefetch((const void*)(bp + (size_t)32 * N), 0, 0);
#pragma unroll
    for (int mt = 0; mt < 4; ++mt) {
      // A fragment: lane row m; two 8-element runs selected by lane half.
      const __bf16* ap = A + (size_t)(m0 + mt * 16 + l15) * K + kk;
      const v8bf a0 = *(const v8bf*)(ap + hl * 8);
      const v8bf a1 = *(const v8bf*)(ap + 16 + hl * 8);
      const v16bf af = cat8(a0, a1);
      acc[mt][0] = __builtin_amdgcn_wmma_f32_16x16x32_bf16(
          false, af, false, bf0, (short)0, acc[mt][0], false, false);
      acc[mt][1] = __builtin_amdgcn_wmma_f32_16x16x32_bf16(
          false, af, false, bf1, (short)0, acc[mt][1], false, false);
    }
  }

  float bval[2] = {0.0f, 0.0f};
  if (bias) { bval[0] = bias[n0 + l15]; bval[1] = bias[n0 + 16 + l15]; }
  float* Cf = (float*)Cout;
  __bf16* Cb = (__bf16*)Cout;
  const int bb2 = m0 / SPAD;          // batch (64-row tiles never cross batches)
#pragma unroll
  for (int mt = 0; mt < 4; ++mt) {
#pragma unroll
    for (int nt = 0; nt < 2; ++nt) {
      const int n = n0 + nt * 16 + l15;
#pragma unroll
      for (int j = 0; j < 8; ++j) {
        const int m = m0 + mt * 16 + j + 8 * hl;
        float val = acc[mt][nt][j] + bval[nt];
        if (resid) val += resid[(size_t)m * N + n];
        if (act == 1) val = 0.5f * val * (1.0f + erff(val * 0.70710678118654752f));
        if (ofmt == 0) {
          Cf[(size_t)m * N + n] = val;
        } else if (ofmt == 1) {
          Cb[(size_t)m * N + n] = (__bf16)val;
        } else {
          const int s = m - bb2 * SPAD;
          Cb[((size_t)bb2 * DM + n) * SPAD + s] = (__bf16)val;
        }
      }
    }
  }
}

// ---------------------------------------------------------------------------
// Embedding: emb = conv1d(circular pad x_enc) + pos_enc + x_mark @ time_w + time_b
// ---------------------------------------------------------------------------
__global__ __launch_bounds__(256) void embed_kernel(
    const float* __restrict__ xe, const float* __restrict__ xm,
    const float* __restrict__ cw, const float* __restrict__ tw,
    const float* __restrict__ tb, float* __restrict__ emb)
{
  const int idx = blockIdx.x * 256 + threadIdx.x;
  if (idx >= BATCH * T_IN * DM) return;
  const int d  = idx & (DM - 1);
  const int bt = idx >> 9;
  const int t  = bt % T_IN;
  const int b  = bt / T_IN;

  float acc = tb[d];
  const float* xmr = xm + ((size_t)b * T_IN + t) * 4;
#pragma unroll
  for (int i = 0; i < 4; ++i) acc += xmr[i] * tw[i * DM + d];

#pragma unroll
  for (int kw = 0; kw < 3; ++kw) {
    int tt = t + kw - 1;
    if (tt < 0) tt = T_IN - 1; else if (tt >= T_IN) tt = 0;
    const float* xr = xe + ((size_t)b * T_IN + tt) * 7;
    const float* wr = cw + (size_t)d * 21 + kw;   // w[d][i][kw], stride 3 over i
#pragma unroll
    for (int i = 0; i < 7; ++i) acc += xr[i] * wr[i * 3];
  }

  const int i2 = d & ~1;
  const float div = __expf(-0.0179889460390160f * (float)i2); // ln(1e4)/512
  const float ang = (float)t * div;
  acc += (d & 1) ? __cosf(ang) : __sinf(ang);
  emb[idx] = acc;
}

// ---------------------------------------------------------------------------
// CSCM stage: strided conv (k=4, stride=4, 128->128) + bias + affine + ELU
// ---------------------------------------------------------------------------
__global__ __launch_bounds__(256) void cscm_kernel(
    const float* __restrict__ in, int in_off, int in_bs,
    float* __restrict__ out, int out_off, int out_bs, int out_len,
    const float* __restrict__ w, const float* __restrict__ cb,
    const float* __restrict__ bg, const float* __restrict__ bb)
{
  const int idx = blockIdx.x * 256 + threadIdx.x;
  if (idx >= BATCH * out_len * DB) return;
  const int c  = idx & (DB - 1);
  const int bt = idx >> 7;
  const int t  = bt % out_len;
  const int b  = bt / out_len;

  float acc = cb[c];
#pragma unroll
  for (int kw = 0; kw < 4; ++kw) {
    const float* ir = in + ((size_t)b * in_bs + in_off + t * 4 + kw) * DB;
    const float* wr = w + ((size_t)c * DB) * 4 + kw;   // w[c][ci][kw]
    for (int ci = 0; ci < DB; ++ci) acc += ir[ci] * wr[ci * 4];
  }
  acc = acc * bg[c] + bb[c];
  acc = (acc > 0.0f) ? acc : (__expf(acc) - 1.0f);
  out[((size_t)b * out_bs + out_off + t) * DB + c] = acc;
}

// ---------------------------------------------------------------------------
// LayerNorm of concat(emb, coarse) rows into seq; pad rows (s>=446) -> 0.
// ---------------------------------------------------------------------------
__global__ __launch_bounds__(32) void ln_build_kernel(
    const float* __restrict__ emb, const float* __restrict__ coarse,
    float* __restrict__ out, const float* __restrict__ g, const float* __restrict__ bb)
{
  const int row  = blockIdx.x;
  const int s    = row % SPAD;
  const int b    = row / SPAD;
  const int lane = threadIdx.x;
  float* orow = out + (size_t)row * DM;
  if (s >= S_VALID) {
#pragma unroll
    for (int i = 0; i < 16; ++i) orow[lane + i * 32] = 0.0f;
    return;
  }
  const float* ir = (s < T_IN) ? (emb + ((size_t)b * T_IN + s) * DM)
                               : (coarse + ((size_t)b * 110 + (s - T_IN)) * DM);
  float vals[16], sum = 0.0f;
#pragma unroll
  for (int i = 0; i < 16; ++i) { vals[i] = ir[lane + i * 32]; sum += vals[i]; }
  const float mu = wave_sum(sum) * (1.0f / 512.0f);
  float vs = 0.0f;
#pragma unroll
  for (int i = 0; i < 16; ++i) { float dd = vals[i] - mu; vs += dd * dd; }
  const float inv = rsqrtf(wave_sum(vs) * (1.0f / 512.0f) + 1e-5f);
#pragma unroll
  for (int i = 0; i < 16; ++i) {
    const int d = lane + i * 32;
    orow[d] = (vals[i] - mu) * inv * g[d] + bb[d];
  }
}

// ---------------------------------------------------------------------------
// LayerNorm rows: out = ln(in) ; pad rows -> 0.
// ---------------------------------------------------------------------------
__global__ __launch_bounds__(32) void ln_rows_kernel(
    const float* __restrict__ in, float* __restrict__ out,
    const float* __restrict__ g, const float* __restrict__ bb)
{
  const int row  = blockIdx.x;
  const int s    = row % SPAD;
  const int lane = threadIdx.x;
  float* orow = out + (size_t)row * DM;
  if (s >= S_VALID) {
#pragma unroll
    for (int i = 0; i < 16; ++i) orow[lane + i * 32] = 0.0f;
    return;
  }
  const float* ir = in + (size_t)row * DM;
  float vals[16], sum = 0.0f;
#pragma unroll
  for (int i = 0; i < 16; ++i) { vals[i] = ir[lane + i * 32]; sum += vals[i]; }
  const float mu = wave_sum(sum) * (1.0f / 512.0f);
  float vs = 0.0f;
#pragma unroll
  for (int i = 0; i < 16; ++i) { float dd = vals[i] - mu; vs += dd * dd; }
  const float inv = rsqrtf(wave_sum(vs) * (1.0f / 512.0f) + 1e-5f);
#pragma unroll
  for (int i = 0; i < 16; ++i) {
    const int d = lane + i * 32;
    orow[d] = (vals[i] - mu) * inv * g[d] + bb[d];
  }
}

// ---------------------------------------------------------------------------
// Fused masked attention for one (b, h, 16-query-row tile), all-bf16 operands.
// q: [b][s][h*64+d] bf16 ; kT: [b][h*64+d][s] bf16 ; v: [b][s][h*64+d] bf16
// scores via WMMA -> LDS f32, softmax, P*V via WMMA, o written bf16.
// grid = (SPAD/16, BATCH*NH), one wave per block.
// ---------------------------------------------------------------------------
__global__ __launch_bounds__(32) void attn_kernel(
    const __bf16* __restrict__ q, const __bf16* __restrict__ kT,
    const __bf16* __restrict__ v, const unsigned char* __restrict__ mask,
    __bf16* __restrict__ o)
{
  __shared__ float sc[16][SPAD + 1];

  const int lane = threadIdx.x;
  const int hl   = lane >> 4;
  const int l15  = lane & 15;
  const int tile = blockIdx.x;
  const int b    = blockIdx.y >> 3;
  const int h    = blockIdx.y & 7;

  const __bf16* qp   = q + ((size_t)b * SPAD + tile * 16 + l15) * DM + h * 64;
  const __bf16* kTb  = kT + ((size_t)b * DM + h * 64) * SPAD;
  const __bf16* vb   = v + ((size_t)b * SPAD) * DM + h * 64;

  // Q fragments for k-chunks 0 and 32 (DK = 64)
  v16bf aq[2];
#pragma unroll
  for (int c = 0; c < 2; ++c) {
    const v8bf a0 = *(const v8bf*)(qp + c * 32 + hl * 8);
    const v8bf a1 = *(const v8bf*)(qp + c * 32 + 16 + hl * 8);
    aq[c] = cat8(a0, a1);
  }

  // ---- scores = Q K^T * scale + mask -> LDS ----
  for (int jt = 0; jt < SPAD / 16; ++jt) {
    v8f s = {};
#pragma unroll
    for (int c = 0; c < 2; ++c) {
      // B = K^T (64x16): lane holds key column l15; k in table order.
      v16bf bk;
#pragma unroll
      for (int e = 0; e < 16; ++e) {
        const int kr = c * 32 + ((e < 8) ? (hl * 8 + e) : (16 + hl * 8 + e - 8));
        bk[e] = kTb[(size_t)kr * SPAD + jt * 16 + l15];
      }
      s = __builtin_amdgcn_wmma_f32_16x16x32_bf16(false, aq[c], false, bk,
                                                  (short)0, s, false, false);
    }
#pragma unroll
    for (int j = 0; j < 8; ++j) {
      const int m  = j + 8 * hl;
      const int ig = tile * 16 + m;
      const int jg = jt * 16 + l15;
      float val = s[j] * 0.125f;  // 1/sqrt(64)
      if (ig >= S_VALID || jg >= S_VALID || mask[ig * S_VALID + jg]) val = -1.0e9f;
      sc[m][jg] = val;
    }
  }
  __syncthreads();

  // ---- softmax per row ----
  for (int r = 0; r < 16; ++r) {
    float mx = -3.0e38f;
    for (int c2 = lane; c2 < SPAD; c2 += 32) mx = fmaxf(mx, sc[r][c2]);
    mx = wave_max(mx);
    float sm = 0.0f;
    for (int c2 = lane; c2 < SPAD; c2 += 32) {
      const float e2 = __expf(sc[r][c2] - mx);
      sc[r][c2] = e2;
      sm += e2;
    }
    sm = wave_sum(sm);
    const float inv = 1.0f / sm;
    for (int c2 = lane; c2 < SPAD; c2 += 32) sc[r][c2] *= inv;
  }
  __syncthreads();

  // ---- O = P * V  (16 x 64) ----
  v8f acc[4] = {};
  for (int kc = 0; kc < SPAD / 32; ++kc) {
    v16bf ap;
#pragma unroll
    for (int e = 0; e < 8; ++e) ap[e]     = (__bf16)sc[l15][kc * 32 + hl * 8 + e];
#pragma unroll
    for (int e = 0; e < 8; ++e) ap[8 + e] = (__bf16)sc[l15][kc * 32 + 16 + hl * 8 + e];
    const __bf16* vr = vb + (size_t)(kc * 32) * DM + l15;
#pragma unroll
    for (int nt = 0; nt < 4; ++nt) {
      v16bf bv;
#pragma unroll
      for (int e = 0; e < 16; ++e) {
        const int kt = (e < 8) ? (hl * 8 + e) : (16 + hl * 8 + e - 8);
        bv[e] = vr[(size_t)kt * DM + nt * 16];
      }
      acc[nt] = __builtin_amdgcn_wmma_f32_16x16x32_bf16(false, ap, false, bv,
                                                        (short)0, acc[nt], false, false);
    }
  }
#pragma unroll
  for (int nt = 0; nt < 4; ++nt) {
#pragma unroll
    for (int j = 0; j < 8; ++j) {
      const int m  = j + 8 * hl;
      const int ig = tile * 16 + m;
      if (ig < S_VALID)
        o[((size_t)b * SPAD + ig) * DM + h * 64 + nt * 16 + l15] = (__bf16)acc[nt][j];
    }
  }
}

// ---------------------------------------------------------------------------
// Final projection
// ---------------------------------------------------------------------------
__global__ __launch_bounds__(256) void pred_kernel(
    const float* __restrict__ seq, const float* __restrict__ pw,
    const int* __restrict__ indexes, float* __restrict__ out)
{
  const int idx = blockIdx.x * 256 + threadIdx.x;
  if (idx >= BATCH * PREDOUT) return;
  const int p = idx % PREDOUT;
  const int b = idx / PREDOUT;
  float acc = 0.0f;
#pragma unroll
  for (int j = 0; j < 4; ++j) {
    const int s = indexes[335 * 4 + j];
    const float* sr = seq + ((size_t)b * SPAD + s) * DM;
    const float* wr = pw + (size_t)(j * DM) * PREDOUT + p;
    for (int d = 0; d < DM; ++d) acc += sr[d] * wr[d * PREDOUT];
  }
  out[idx] = acc;
}

// ---------------------------------------------------------------------------
// Host orchestration
// ---------------------------------------------------------------------------
extern "C" void kernel_launch(void* const* d_in, const int* in_sizes, int n_in,
                              void* d_out, int out_size, void* d_ws, size_t ws_size,
                              hipStream_t stream)
{
  const float* x_enc   = (const float*)d_in[0];
  const float* x_mark  = (const float*)d_in[1];
  const float* conv_w  = (const float*)d_in[4];
  const float* time_w  = (const float*)d_in[5];
  const float* time_b  = (const float*)d_in[6];
  const float* down_w  = (const float*)d_in[7];
  const float* down_b  = (const float*)d_in[8];
  const float* cscm_w  = (const float*)d_in[9];
  const float* cscm_b  = (const float*)d_in[10];
  const float* bn_g    = (const float*)d_in[11];
  const float* bn_b    = (const float*)d_in[12];
  const float* up_w    = (const float*)d_in[13];
  const float* up_b    = (const float*)d_in[14];
  const float* lng     = (const float*)d_in[15];
  const float* lnb     = (const float*)d_in[16];
  const float* wq      = (const float*)d_in[17];
  const float* wk      = (const float*)d_in[18];
  const float* wv      = (const float*)d_in[19];
  const float* wo      = (const float*)d_in[20];
  const float* ln1g    = (const float*)d_in[21];
  const float* ln1b    = (const float*)d_in[22];
  const float* w1      = (const float*)d_in[23];
  const float* b1      = (const float*)d_in[24];
  const float* w2      = (const float*)d_in[25];
  const float* b2      = (const float*)d_in[26];
  const float* ln2g    = (const float*)d_in[27];
  const float* ln2b    = (const float*)d_in[28];
  const float* pred_w  = (const float*)d_in[29];
  const unsigned char* mask = (const unsigned char*)d_in[30];
  const int*   indexes = (const int*)d_in[31];
  float* out = (float*)d_out;

  // ---- workspace layout (bytes) ----
  char* ws = (char*)d_ws;
  const size_t SEQ_BYTES = (size_t)BATCH * SPAD * DM * 4;     // 29,360,128
  const size_t BF_BYTES  = (size_t)BATCH * SPAD * DM * 2;     // 14,680,064
  float*  seq    = (float*)(ws);
  float*  tmp    = (float*)(ws + SEQ_BYTES);
  __bf16* q_bf   = (__bf16*)(ws + 2 * SEQ_BYTES);
  __bf16* kT_bf  = (__bf16*)(ws + 2 * SEQ_BYTES + BF_BYTES);
  __bf16* v_bf   = (__bf16*)(ws + 2 * SEQ_BYTES + 2 * BF_BYTES);
  __bf16* seq_bf = (__bf16*)(ws + 2 * SEQ_BYTES + 3 * BF_BYTES);
  __bf16* wbf    = (__bf16*)(ws + 2 * SEQ_BYTES + 4 * BF_BYTES);
  __bf16* h_bf   = q_bf;   // FFN hidden chunk (7168 x 2048 bf16) = q+kT regions

  // bf16 weights inside wbf
  __bf16* wq_bf = wbf;                       // 3*512*512
  __bf16* wk_bf = wq_bf + 786432;
  __bf16* wv_bf = wk_bf + 786432;
  __bf16* wo_bf = wv_bf + 786432;
  __bf16* w1_bf = wo_bf + 786432;            // 3*512*2048
  __bf16* w2_bf = w1_bf + 3145728;
  __bf16* dw_bf = w2_bf + 3145728;           // 512*128
  __bf16* uw_bf = dw_bf + 65536;             // 128*512

  auto cvt = [&](const float* s, __bf16* d, int n) {
    const int n4 = n / 4;
    cvt_bf16_kernel<<<(n4 + 255) / 256, 256, 0, stream>>>(s, d, n4);
  };
  auto gemm = [&](const __bf16* A, const __bf16* Bw, const float* bias,
                  const float* resid, void* C, int M, int N, int K,
                  int act, int ofmt) {
    dim3 grid(N / 128, M / 64), block(32, 4);
    gemm_bf16_kernel<<<grid, block, 0, stream>>>(A, Bw, bias, resid, C,
                                                 M, N, K, act, ofmt);
  };

  // ---- weight conversion ----
  cvt(wq, wq_bf, 786432);
  cvt(wk, wk_bf, 786432);
  cvt(wv, wv_bf, 786432);
  cvt(wo, wo_bf, 786432);
  cvt(w1, w1_bf, 3145728);
  cvt(w2, w2_bf, 3145728);
  cvt(down_w, dw_bf, 65536);
  cvt(up_w, uw_bf, 65536);

  // ---- Embedding + CSCM pyramid ----
  float*  emb    = tmp;                 // 32*336*512 f32
  __bf16* emb_bf = seq_bf;              // 32*336*512 bf16
  float*  x0     = (float*)q_bf;        // 32*336*128 f32 (q region)
  float*  xc     = (float*)v_bf;        // 32*110*128 f32 (v region)
  __bf16* xc_bf  = kT_bf;               // 32*110*128 bf16
  float*  coarse = (float*)v_bf;        // 32*110*512 f32 (xc dead by then)

  embed_kernel<<<(BATCH * T_IN * DM) / 256, 256, 0, stream>>>(
      x_enc, x_mark, conv_w, time_w, time_b, emb);
  cvt(emb, emb_bf, BATCH * T_IN * DM);
  gemm(emb_bf, dw_bf, down_b, nullptr, x0, BATCH * T_IN, DB, DM, 0, 0);

  cscm_kernel<<<(BATCH * 84 * DB) / 256, 256, 0, stream>>>(
      x0, 0, T_IN, xc, 0, 110, 84,
      cscm_w + 0 * DB * DB * 4, cscm_b + 0 * DB, bn_g + 0 * DB, bn_b + 0 * DB);
  cscm_kernel<<<(BATCH * 21 * DB) / 256, 256, 0, stream>>>(
      xc, 0, 110, xc, 84, 110, 21,
      cscm_w + 1 * DB * DB * 4, cscm_b + 1 * DB, bn_g + 1 * DB, bn_b + 1 * DB);
  cscm_kernel<<<(BATCH * 5 * DB) / 256, 256, 0, stream>>>(
      xc, 84, 110, xc, 105, 110, 5,
      cscm_w + 2 * DB * DB * 4, cscm_b + 2 * DB, bn_g + 2 * DB, bn_b + 2 * DB);

  cvt(xc, xc_bf, BATCH * 110 * DB);
  gemm(xc_bf, uw_bf, up_b, nullptr, coarse, BATCH * 110, DM, DB, 0, 0);

  ln_build_kernel<<<BATCH * SPAD, 32, 0, stream>>>(emb, coarse, seq, lng, lnb);

  // ---- Encoder layers ----
  const int MROWS = BATCH * SPAD;          // 14336
  for (int l = 0; l < 3; ++l) {
    const size_t ws512 = (size_t)l * DM * DM;
    cvt(seq, seq_bf, MROWS * DM);
    gemm(seq_bf, wq_bf + ws512, nullptr, nullptr, q_bf,  MROWS, DM, DM, 0, 1);
    gemm(seq_bf, wk_bf + ws512, nullptr, nullptr, kT_bf, MROWS, DM, DM, 0, 2);
    gemm(seq_bf, wv_bf + ws512, nullptr, nullptr, v_bf,  MROWS, DM, DM, 0, 1);

    attn_kernel<<<dim3(SPAD / 16, BATCH * NH), 32, 0, stream>>>(
        q_bf, kT_bf, v_bf, mask, q_bf /* o aliases q */);

    gemm(q_bf, wo_bf + ws512, nullptr, seq, tmp, MROWS, DM, DM, 0, 0);
    ln_rows_kernel<<<MROWS, 32, 0, stream>>>(tmp, seq, ln1g + l * DM, ln1b + l * DM);

    cvt(seq, seq_bf, MROWS * DM);
    for (int c = 0; c < 2; ++c) {
      const int off = c * (MROWS / 2);   // 7168 rows / chunk
      gemm(seq_bf + (size_t)off * DM, w1_bf + (size_t)l * DM * DFF, b1 + l * DFF,
           nullptr, h_bf, MROWS / 2, DFF, DM, 1, 1);
      gemm(h_bf, w2_bf + (size_t)l * DFF * DM, b2 + l * DM,
           seq + (size_t)off * DM, tmp + (size_t)off * DM, MROWS / 2, DM, DFF, 0, 0);
    }
    ln_rows_kernel<<<MROWS, 32, 0, stream>>>(tmp, seq, ln2g + l * DM, ln2b + l * DM);
  }

  // ---- Final gather + projection ----
  pred_kernel<<<(BATCH * PREDOUT + 255) / 256, 256, 0, stream>>>(
      seq, pred_w, indexes, out);
}
